// ComponentNN_30124900614672
// MI455X (gfx1250) — matrix operand order repaired and checked by reading, not compiled, not run on hardware
//
#include <hip/hip_runtime.h>
#include <cmath>

// CDNA5 / gfx1250: wave32, WMMA 16x16x32 bf16 (f32 acc), async global->LDS staging.
typedef __attribute__((ext_vector_type(16))) __bf16 v16bf;
typedef __attribute__((ext_vector_type(8)))  float  v8f;

union FragBF {
    v16bf v;
    uint4 q[2];   // two 16-byte halves
};

// ---------------------------------------------------------------------------
// Elementwise f32 -> (bf16 hi, bf16 lo) split:  x ~= hi + lo
// ---------------------------------------------------------------------------
__global__ void split_bf16_kernel(const float* __restrict__ in,
                                  __bf16* __restrict__ hi,
                                  __bf16* __restrict__ lo, int n) {
    int i = blockIdx.x * blockDim.x + threadIdx.x;
    int stride = gridDim.x * blockDim.x;
    for (; i < n; i += stride) {
        float x = in[i];
        __bf16 h = (__bf16)x;
        hi[i] = h;
        lo[i] = (__bf16)(x - (float)h);
    }
}

// ---------------------------------------------------------------------------
// Blended-expert layer, LDS-shared weights.
//   out[b,o] = sum_e blend[e,b] * ( sum_i w[e,o,i]*h[b,i] + bias[e,o] )
//
// Block = 256 threads = 8 waves. All waves share ONE 16-wide out-feature tile;
// each wave owns a different 16-row batch tile (128 rows / block).
// Per K-step(32): block stages W[e, ocol..+15, k..k+31] (8 experts, hi+lo,
// 16KB -> 20KB padded) into LDS via global_load_async_to_lds_b128 (ASYNCcnt),
// double-buffered; each wave then builds B frags from LDS and its own A frags
// from global, issuing 3 WMMAs per expert (bf16 split ~ f32 accuracy).
// ---------------------------------------------------------------------------
#define ROWB   80          // padded LDS row stride (64B data + 16B pad)
#define ARRB   (16 * ROWB) // one (expert, hi/lo) slab: 16 rows
#define BUFB   (16 * ARRB) // one k-step buffer: 8 experts * 2 slabs = 20480 B

template <int D_IN, bool ACT, bool WRITE_F32>
__global__ __launch_bounds__(256)
void blend_layer_kernel(const __bf16* __restrict__ Hhi,
                        const __bf16* __restrict__ Hlo,
                        const __bf16* __restrict__ Whi,
                        const __bf16* __restrict__ Wlo,
                        const float*  __restrict__ bias,   // [E, d_out]
                        const float*  __restrict__ blend,  // [E, 1024]
                        float*  __restrict__ outF32,       // [1024, d_out] (WRITE_F32)
                        __bf16* __restrict__ outHi,        // else: split for next layer
                        __bf16* __restrict__ outLo,
                        int d_out) {
    constexpr int BATCH = 1024;
    constexpr int E = 8;

    __shared__ __align__(16) char smem[2 * BUFB];   // 40 KB double buffer

    const int tid  = threadIdx.x;
    const int wave = tid >> 5;
    const int lane = tid & 31;
    const int half = (lane >> 4) & 1;   // lane group 0..15 / 16..31
    const int l15  = lane & 15;

    const int brow = blockIdx.y * 128 + wave * 16;  // this wave's batch tile (M)
    const int ocol = blockIdx.x * 16;               // block's out-feature tile (N)

    // ISA 16-bit fragment layouts (CDNA5 ISA §7.12.2):
    const int a_k0    = half * 8;       // A: lanes 16-31 hold K=8..15,24..31
    const int row_off = half * 8;       // C/D: lanes 16-31 hold M=8..15
    // B: lanes 16-31 hold K=8..15,24..31 -> byte offsets half*16 and 32+half*16 within a 64B row

    // A fragment source: activation row (brow + l15), row-major [1024, D_IN]
    const __bf16* Ahi = Hhi + (size_t)(brow + l15) * D_IN;
    const __bf16* Alo = Hlo + (size_t)(brow + l15) * D_IN;

    const uint32_t smem_base = (uint32_t)(uintptr_t)&smem[0];

    // Stage one k-step of weights (all experts, hi+lo) into LDS buffer `buf`.
    // 1024 16-byte chunks; 4 per thread.
    auto stage = [&](int k, int buf) {
#pragma unroll
        for (int p = 0; p < 4; ++p) {
            const int c   = tid + p * 256;
            const int arr = c >> 6;          // 0..15: (expert, hi/lo)
            const int r   = (c >> 2) & 15;   // weight row within tile
            const int seg = c & 3;           // 16B segment within 64B of K
            const int e   = arr >> 1;
            const int hl  = arr & 1;
            int o = ocol + r;
            if (o >= d_out) o = d_out - 1;   // clamp partial N tile
            const __bf16* g = (hl ? Wlo : Whi) +
                              ((size_t)e * d_out + o) * D_IN + k + seg * 8;
            const uint32_t lds = smem_base + buf * BUFB + arr * ARRB + r * ROWB + seg * 16;
            const uint64_t ga  = (uint64_t)(uintptr_t)g;
            asm volatile("global_load_async_to_lds_b128 %0, %1, off"
                         :: "v"(lds), "v"(ga) : "memory");
        }
    };

    // B column n == weight row o in LDS slab
    int o = ocol + l15;
    if (o >= d_out) o = d_out - 1;

    v8f acc[E];
    const v8f vzero = {0.f, 0.f, 0.f, 0.f, 0.f, 0.f, 0.f, 0.f};
#pragma unroll
    for (int e = 0; e < E; ++e) acc[e] = vzero;

    int buf = 0;
    stage(0, 0);
    asm volatile("s_wait_asynccnt 0" ::: "memory");
    __syncthreads();

    for (int k = 0; k < D_IN; k += 32) {
        if (k + 32 < D_IN) stage(k + 32, buf ^ 1);

        FragBF ahi, alo;
        ahi.q[0] = *(const uint4*)(Ahi + k + a_k0);
        ahi.q[1] = *(const uint4*)(Ahi + k + a_k0 + 16);
        alo.q[0] = *(const uint4*)(Alo + k + a_k0);
        alo.q[1] = *(const uint4*)(Alo + k + a_k0 + 16);

        const char* base = smem + buf * BUFB + l15 * ROWB + half * 16;
#pragma unroll
        for (int e = 0; e < E; ++e) {
            const char* bh = base + (e * 2 + 0) * ARRB;
            const char* bl = base + (e * 2 + 1) * ARRB;
            FragBF bhi, blo;
            bhi.q[0] = *(const uint4*)(bh);
            bhi.q[1] = *(const uint4*)(bh + 32);
            blo.q[0] = *(const uint4*)(bl);
            blo.q[1] = *(const uint4*)(bl + 32);
            // D = A*B + C ; 3-product bf16 split ~ f32 accuracy
            acc[e] = __builtin_amdgcn_wmma_f32_16x16x32_bf16(
                false, ahi.v, false, bhi.v, (short)0, acc[e], false, false);
            acc[e] = __builtin_amdgcn_wmma_f32_16x16x32_bf16(
                false, ahi.v, false, blo.v, (short)0, acc[e], false, false);
            acc[e] = __builtin_amdgcn_wmma_f32_16x16x32_bf16(
                false, alo.v, false, bhi.v, (short)0, acc[e], false, false);
        }

        asm volatile("s_wait_asynccnt 0" ::: "memory");
        __syncthreads();
        buf ^= 1;
    }

    // Epilogue: per-row blend coefficients fold the expert sum + bias.
    float out[8] = {0.f, 0.f, 0.f, 0.f, 0.f, 0.f, 0.f, 0.f};
#pragma unroll
    for (int e = 0; e < E; ++e) {
        const float be = bias[(size_t)e * d_out + o];
        const float* blv = blend + (size_t)e * BATCH + brow + row_off;
#pragma unroll
        for (int r = 0; r < 8; ++r)
            out[r] += blv[r] * (acc[e][r] + be);
    }

    const int col  = ocol + l15;
    const bool cok = col < d_out;
#pragma unroll
    for (int r = 0; r < 8; ++r) {
        float v = out[r];
        if (ACT) v = (v > 0.f) ? v : expm1f(v);   // ELU(alpha=1)
        const size_t idx = (size_t)(brow + row_off + r) * d_out + col;
        if (cok) {
            if (WRITE_F32) {
                outF32[idx] = v;
            } else {
                __bf16 h = (__bf16)v;
                outHi[idx] = h;
                outLo[idx] = (__bf16)(v - (float)h);
            }
        }
    }
}

// ---------------------------------------------------------------------------
extern "C" void kernel_launch(void* const* d_in, const int* in_sizes, int n_in,
                              void* d_out, int out_size, void* d_ws, size_t ws_size,
                              hipStream_t stream) {
    (void)in_sizes; (void)n_in; (void)out_size; (void)ws_size;

    const float* x  = (const float*)d_in[0];  // [1024, 480]
    const float* wb = (const float*)d_in[1];  // [8, 1024]
    const float* w0 = (const float*)d_in[2];  // [8, 512, 480]
    const float* b0 = (const float*)d_in[3];  // [8, 512]
    const float* w1 = (const float*)d_in[4];  // [8, 512, 512]
    const float* b1 = (const float*)d_in[5];  // [8, 512]
    const float* w2 = (const float*)d_in[6];  // [8, 311, 512]
    const float* b2 = (const float*)d_in[7];  // [8, 311]
    float* out = (float*)d_out;               // [1024, 311]

    constexpr int B = 1024, E = 8;
    constexpr int d0 = 480, d1 = 512, d2 = 512, d3 = 311;
    const size_t nW0 = (size_t)E * d1 * d0;
    const size_t nW1 = (size_t)E * d2 * d1;
    const size_t nW2 = (size_t)E * d3 * d2;
    const size_t nX  = (size_t)B * d0;
    const size_t nH1 = (size_t)B * d1;
    const size_t nH2 = (size_t)B * d2;

    // Workspace carve-out (256B aligned sub-buffers), ~27.5 MB total.
    char* ws = (char*)d_ws;
    size_t off = 0;
    auto alloc_bf = [&](size_t n) {
        __bf16* p = (__bf16*)(ws + off);
        off = (off + n * 2 + 255) & ~(size_t)255;
        return p;
    };
    __bf16* W0hi = alloc_bf(nW0); __bf16* W0lo = alloc_bf(nW0);
    __bf16* W1hi = alloc_bf(nW1); __bf16* W1lo = alloc_bf(nW1);
    __bf16* W2hi = alloc_bf(nW2); __bf16* W2lo = alloc_bf(nW2);
    __bf16* Xhi  = alloc_bf(nX);  __bf16* Xlo  = alloc_bf(nX);
    __bf16* H1hi = alloc_bf(nH1); __bf16* H1lo = alloc_bf(nH1);
    __bf16* H2hi = alloc_bf(nH2); __bf16* H2lo = alloc_bf(nH2);

    auto split = [&](const float* src, __bf16* hi, __bf16* lo, size_t n) {
        int blocks = (int)((n + 255) / 256);
        split_bf16_kernel<<<blocks, 256, 0, stream>>>(src, hi, lo, (int)n);
    };
    split(w0, W0hi, W0lo, nW0);
    split(w1, W1hi, W1lo, nW1);
    split(w2, W2hi, W2lo, nW2);
    split(x,  Xhi,  Xlo,  nX);

    // Layer 0: [1024,480] x [8,512,480] -> ELU -> bf16 split
    dim3 g0(d1 / 16, B / 128);
    blend_layer_kernel<d0, true, false><<<g0, 256, 0, stream>>>(
        Xhi, Xlo, W0hi, W0lo, b0, wb, nullptr, H1hi, H1lo, d1);

    // Layer 1: [1024,512] x [8,512,512] -> ELU -> bf16 split
    dim3 g1(d2 / 16, B / 128);
    blend_layer_kernel<d1, true, false><<<g1, 256, 0, stream>>>(
        H1hi, H1lo, W1hi, W1lo, b1, wb, nullptr, H2hi, H2lo, d2);

    // Layer 2: [1024,512] x [8,311,512] -> f32 output (no activation)
    dim3 g2((d3 + 15) / 16, B / 128);
    blend_layer_kernel<d2, false, true><<<g2, 256, 0, stream>>>(
        H2hi, H2lo, W2hi, W2lo, b2, wb, out, nullptr, nullptr, d3);
}
